// Proposer_26714696581169
// MI455X (gfx1250) — matrix operand order repaired and checked by reading, not compiled, not run on hardware
//
#include <hip/hip_runtime.h>
#include <hip/hip_bf16.h>
#include <math.h>

typedef __attribute__((ext_vector_type(16))) __bf16 v16bf;
typedef __attribute__((ext_vector_type(8)))  float  v8f;
typedef __attribute__((ext_vector_type(4)))  unsigned int u32x4;
typedef __attribute__((ext_vector_type(8)))  int i32x8;
typedef __attribute__((ext_vector_type(4)))  int i32x4;

#define H     512
#define LSEQ  4096
#define NB    8
#define NKER  4
#define NEGF  (-100000000.0f)

#define WM 2   // 16-row WMMA subtiles per wave (32 rows)
#define WN 4   // 16-col WMMA subtiles per wave (64 cols)

union Frag { v16bf v; uint4 u[2]; };

__device__ __forceinline__ unsigned short f2bf(float f) {
  unsigned int u = __float_as_uint(f);
  unsigned int r = u + 0x7FFFu + ((u >> 16) & 1u);   // round-to-nearest-even
  return (unsigned short)(r >> 16);
}
__device__ __forceinline__ float bf2f(unsigned short h) {
  return __uint_as_float(((unsigned int)h) << 16);
}
__device__ __forceinline__ float gelu_tanh(float x) {
  float x3 = x * x * x;
  return 0.5f * x * (1.0f + tanhf(0.7978845608028654f * (x + 0.044715f * x3)));
}

// ---------------------------------------------------------------------------
// fp32 -> bf16 elementwise convert
// ---------------------------------------------------------------------------
__global__ void f32_to_bf16_kernel(const float* __restrict__ s,
                                   unsigned short* __restrict__ d, long long n) {
  long long i = (long long)blockIdx.x * blockDim.x + threadIdx.x;
  if (i < n) d[i] = f2bf(s[i]);
}

// ---------------------------------------------------------------------------
// W (K,N) fp32 row-major -> Wt (Npad,K) bf16, zero-pad cols n >= N
// ---------------------------------------------------------------------------
__global__ void transpose_w_kernel(const float* __restrict__ W,
                                   unsigned short* __restrict__ Wt,
                                   int K, int N, int Npad) {
  long long i = (long long)blockIdx.x * blockDim.x + threadIdx.x;
  if (i >= (long long)Npad * K) return;
  int n = (int)(i / K);
  int k = (int)(i - (long long)n * K);
  float v = (n < N) ? W[(long long)k * N + n] : 0.0f;
  Wt[i] = f2bf(v);
}

// ---------------------------------------------------------------------------
// Generic WMMA GEMM: D = act(A(bf16, MxK) @ Bt(bf16, stored (Npad,K)) + bias)
// 32x64 macro-tile per wave (2x4 WMMA subtiles), 8 waves per block.
// Conv mode (Lvalid>0): row m -> b=m/Lvalid, p=m%Lvalid, A row base (b*Ltot+p)*lda
// (overlapped rows give the k-tap conv since taps are contiguous in memory).
// ---------------------------------------------------------------------------
struct GemmArgs {
  const unsigned short* A;
  const unsigned short* Bw;   // (Npad, K) bf16 transposed weights
  const float* bias;          // length >= Nout
  unsigned short* outB;       // optional bf16 output (ldOutB), or null
  float* outF;                // optional fp32 output (ldOutF), or null
  int M, Npad, K;
  int lda;
  int Lvalid, Ltot;           // Lvalid>0 => conv row mapping
  int ldOutB, ldOutF;
  int Nout;                   // valid columns for bias+store (<= Npad)
  int act;                    // 1 = gelu
};

__global__ __launch_bounds__(256) void wmma_gemm_kernel(GemmArgs g) {
  const int lane = threadIdx.x & 31;
  const int wave = threadIdx.x >> 5;
  const int ntiles = (g.Npad + 16 * WN - 1) / (16 * WN);
  const int mtiles = (g.M + 16 * WM - 1) / (16 * WM);
  long long tile = (long long)blockIdx.x * 8 + wave;
  if (tile >= (long long)mtiles * ntiles) return;
  const int mt = (int)(tile / ntiles);
  const int nt = (int)(tile - (long long)mt * ntiles);

  const int r  = lane & 15;
  const int hi = lane >> 4;

  // A row bases (ISA 16-bit A layout: both lane halves cover rows 0..15; hi
  // selects the K-half).
  long long abase[WM];
#pragma unroll
  for (int i = 0; i < WM; ++i) {
    int row  = mt * (16 * WM) + i * 16 + r;
    int rowc = (row < g.M) ? row : (g.M - 1);
    if (g.Lvalid > 0) {
      int b = rowc / g.Lvalid, p = rowc - b * g.Lvalid;
      abase[i] = ((long long)b * g.Ltot + p) * (long long)g.lda;
    } else {
      abase[i] = (long long)rowc * g.lda;
    }
  }
  // B column bases; Bt stored (Npad, K); each half loads two 8-value K runs
  // (k0+hi*8 and k0+16+hi*8), matching A.
  long long bbase[WN];
  int ncol[WN];
#pragma unroll
  for (int j = 0; j < WN; ++j) {
    int n   = nt * (16 * WN) + j * 16 + r;
    ncol[j] = n;
    int nc  = (n < g.Npad) ? n : (g.Npad - 1);
    bbase[j] = (long long)nc * g.K;
  }

  v8f c[WM][WN];
#pragma unroll
  for (int i = 0; i < WM; ++i)
#pragma unroll
    for (int j = 0; j < WN; ++j)
      c[i][j] = (v8f){0.f, 0.f, 0.f, 0.f, 0.f, 0.f, 0.f, 0.f};

  for (int k0 = 0; k0 < g.K; k0 += 32) {
    Frag af[WM], bfr[WN];
    // Issue all 12 b128 loads up front; distinct fragment registers let the
    // scheduler wait incrementally instead of s_wait_loadcnt 0 per WMMA pair.
#pragma unroll
    for (int i = 0; i < WM; ++i) {
      af[i].u[0] = *reinterpret_cast<const uint4*>(g.A + abase[i] + k0 + hi * 8);
      af[i].u[1] = *reinterpret_cast<const uint4*>(g.A + abase[i] + k0 + 16 + hi * 8);
    }
#pragma unroll
    for (int j = 0; j < WN; ++j) {
      bfr[j].u[0] = *reinterpret_cast<const uint4*>(g.Bw + bbase[j] + k0 + hi * 8);
      bfr[j].u[1] = *reinterpret_cast<const uint4*>(g.Bw + bbase[j] + k0 + 16 + hi * 8);
    }
    if (k0 + 32 < g.K) {  // near-cache prefetch of next K-chunk of the A stream
      __builtin_prefetch(g.A + abase[0] + k0 + 32, 0, 3);
    }
#pragma unroll
    for (int j = 0; j < WN; ++j) {
#pragma unroll
      for (int i = 0; i < WM; ++i) {
        c[i][j] = __builtin_amdgcn_wmma_f32_16x16x32_bf16(false, af[i].v, false, bfr[j].v,
                                                          (short)0, c[i][j], false, false);
      }
    }
  }

  // Epilogue: bias (one column per lane in C/D layout) + activation + stores.
#pragma unroll
  for (int j = 0; j < WN; ++j) {
    int n = ncol[j];
    if (n >= g.Nout) continue;
    float bv = (g.bias != nullptr) ? g.bias[n] : 0.0f;
#pragma unroll
    for (int i = 0; i < WM; ++i) {
#pragma unroll
      for (int v = 0; v < 8; ++v) {
        float x = c[i][j][v] + bv;
        x = g.act ? gelu_tanh(x) : x;
        int orow = mt * (16 * WM) + i * 16 + hi * 8 + v;
        if (orow < g.M) {
          long long roff;
          if (g.Lvalid > 0) {
            int b = orow / g.Lvalid, p = orow - b * g.Lvalid;
            roff = (long long)b * g.Ltot + p;
          } else {
            roff = orow;
          }
          if (g.outB) g.outB[roff * g.ldOutB + n] = f2bf(x);
          if (g.outF) g.outF[roff * g.ldOutF + n] = x;
        }
      }
    }
  }
}

// ---------------------------------------------------------------------------
// scores = h1(bf16, rows x 256) @ attn_w2 (256x2) + attn_b2  (N=2: VALU dot).
// attn_w2 (2 KB) is staged into LDS by the Tensor Data Mover: wave 0 builds a
// D# descriptor (2-D tensor, data_size=4B, tile 512x1), issues
// tensor_load_to_lds, waits on TENSORcnt, then the block synchronizes.
// ---------------------------------------------------------------------------
__global__ __launch_bounds__(256) void scores_kernel(
    const unsigned short* __restrict__ h1, const float* __restrict__ w2,
    const float* __restrict__ b2, float* __restrict__ sc, int Mrows) {
  __shared__ float w2s[512];

  if (threadIdx.x < 32) {   // one wave issues the TDM descriptor
    unsigned long long ga = (unsigned long long)(uintptr_t)w2;
    unsigned int lds = (unsigned int)(uintptr_t)&w2s[0];  // low 32b = LDS offset

    u32x4 g0;
    g0[0] = 1u;                                              // count = 1 (valid D#)
    g0[1] = lds;                                             // lds_addr
    g0[2] = (unsigned int)(ga & 0xFFFFFFFFu);                // global_addr[31:0]
    g0[3] = (unsigned int)((ga >> 32) & 0x1FFFFFFu)          // global_addr[56:32]
            | (2u << 30);                                    // type = 2 ("image")

    i32x8 g1;
    g1[0] = (int)(2u << 16);        // workgroup_mask=0, data_size=2 (4 bytes)
    g1[1] = (int)(512u << 16);      // tensor_dim0[15:0]=512 (bits 63:48)
    g1[2] = (int)(1u << 16);        // tensor_dim0 hi=0; tensor_dim1[15:0]=1
    g1[3] = (int)(512u << 16);      // tensor_dim1 hi=0; tile_dim0=512
    g1[4] = 1;                      // tile_dim1=1; tile_dim2=0
    g1[5] = 512;                    // tensor_dim0_stride[31:0]=512
    g1[6] = (int)(512u << 16);      // dim0_stride hi=0; tensor_dim1_stride lo=512
    g1[7] = 0;                      // tensor_dim1_stride hi=0

    i32x4 gz = {0, 0, 0, 0};        // groups 2/3: dims 2+ unused (tile_dim*=0)
#if defined(__clang_major__) && (__clang_major__ >= 23)
    i32x8 gz8 = {0, 0, 0, 0, 0, 0, 0, 0};
    __builtin_amdgcn_tensor_load_to_lds(g0, g1, gz, gz, gz8, 0);
#else
    __builtin_amdgcn_tensor_load_to_lds(g0, g1, gz, gz, 0);
#endif
    __builtin_amdgcn_s_wait_tensorcnt(0);
  }
  __syncthreads();

  int m = blockIdx.x * blockDim.x + threadIdx.x;
  if (m >= Mrows) return;
  const unsigned short* rowp = h1 + (long long)m * 256;
  float s0 = b2[0], s1 = b2[1];
#pragma unroll 4
  for (int i = 0; i < 256; ++i) {
    float v = bf2f(rowp[i]);
    s0 += v * w2s[2 * i];
    s1 += v * w2s[2 * i + 1];
  }
  sc[2 * (long long)m]     = s0;
  sc[2 * (long long)m + 1] = s1;
}

// ---------------------------------------------------------------------------
// Gather + mask + softmax over span candidates + weighted span endpoints.
// scores stored as (j, b, q, 2); one thread per (b, p).
// ---------------------------------------------------------------------------
__global__ __launch_bounds__(256) void gather_kernel(
    const float* __restrict__ sc, const unsigned char* __restrict__ masks,
    const int* __restrict__ sel, float* __restrict__ outlocs, int smax) {
  int t = blockIdx.x * blockDim.x + threadIdx.x;
  if (t >= NB * LSEQ) return;
  int b = t / LSEQ, p = t - b * LSEQ;
  const int offs[NKER] = {0, 1, 3, 6};

  float s0[32], s1[32], l0[32], l1[32];
  float m0 = NEGF, m1 = NEGF;
  for (int s = 0; s < smax; ++s) {
    int idx = sel[(long long)p * smax + s];
    float v0 = NEGF, v1 = NEGF, g0 = 0.f, g1 = 0.f;
    if (idx >= 0) {
      int j = idx >> 12;        // idx / 4096
      int q = idx & 4095;
      g0 = (float)q;
      g1 = (float)(q + offs[j]);
      bool valid = (masks[(long long)b * LSEQ + q] != 0) && (q + offs[j] < LSEQ);
      if (valid) {
        long long si = (((long long)j * NB + b) * LSEQ + q) * 2;
        v0 = sc[si];
        v1 = sc[si + 1];
      }
    }
    s0[s] = v0; s1[s] = v1; l0[s] = g0; l1[s] = g1;
    m0 = fmaxf(m0, v0); m1 = fmaxf(m1, v1);
  }
  float e0 = 0.f, e1 = 0.f, a0 = 0.f, a1 = 0.f;
  for (int s = 0; s < smax; ++s) {
    float w0 = __expf(s0[s] - m0), w1 = __expf(s1[s] - m1);
    e0 += w0; e1 += w1;
    a0 += w0 * l0[s]; a1 += w1 * l1[s];
  }
  outlocs[2 * (long long)t]     = a0 / e0;
  outlocs[2 * (long long)t + 1] = a1 / e1;
}

// ---------------------------------------------------------------------------
extern "C" void kernel_launch(void* const* d_in, const int* in_sizes, int n_in,
                              void* d_out, int out_size, void* d_ws, size_t ws_size,
                              hipStream_t stream) {
  (void)n_in; (void)out_size; (void)ws_size;
  const float*         x      = (const float*)d_in[0];
  const unsigned char* masks  = (const unsigned char*)d_in[1];
  const int*           sel    = (const int*)d_in[2];
  const float*         convW[NKER] = {(const float*)d_in[3], (const float*)d_in[4],
                                      (const float*)d_in[5], (const float*)d_in[6]};
  const float*         convB[NKER] = {(const float*)d_in[7], (const float*)d_in[8],
                                      (const float*)d_in[9], (const float*)d_in[10]};
  const float* attn_w1 = (const float*)d_in[11];
  const float* attn_b1 = (const float*)d_in[12];
  const float* attn_w2 = (const float*)d_in[13];
  const float* attn_b2 = (const float*)d_in[14];
  const float* cls_w   = (const float*)d_in[15];
  const float* cls_b   = (const float*)d_in[16];

  int smax = in_sizes[2] / LSEQ;
  if (smax > 32) smax = 32;

  const int KER[NKER]  = {1, 2, 3, 4};
  const int OFFS[NKER] = {0, 1, 3, 6};
  const int TYPESP1 = 101, NPAD_CLS = 112, HH = H / 2;
  const long long PLANE = (long long)NB * LSEQ * H;   // 16,777,216

  // d_out layout: queries | queries_locs | queries_logits
  float* out_queries = (float*)d_out;
  float* out_locs    = out_queries + PLANE;
  float* out_logits  = out_locs + (long long)NB * LSEQ * 2;

  // Workspace layout (all offsets 256B-aligned).
  char* ws = (char*)d_ws;
  unsigned short* xbf   = (unsigned short*)ws;                         ws += PLANE * 2;
  unsigned short* feats = (unsigned short*)ws;                         ws += NKER * PLANE * 2;
  unsigned short* h1    = (unsigned short*)ws;                         ws += (long long)NKER * NB * LSEQ * HH * 2;
  float*          scbuf = (float*)ws;                                  ws += (long long)NKER * NB * LSEQ * 2 * 4;
  unsigned short* convWt[NKER];
  for (int j = 0; j < NKER; ++j) { convWt[j] = (unsigned short*)ws;    ws += (long long)H * (KER[j] * H) * 2; }
  unsigned short* attnWt = (unsigned short*)ws;                        ws += (long long)HH * H * 2;
  unsigned short* clsWt  = (unsigned short*)ws;                        ws += (long long)NPAD_CLS * H * 2;

  // 1) Convert activations input to bf16.
  {
    long long n = PLANE;
    f32_to_bf16_kernel<<<(unsigned)((n + 255) / 256), 256, 0, stream>>>(x, xbf, n);
  }
  // 2) Transpose+convert weights to (Npad, K) bf16.
  for (int j = 0; j < NKER; ++j) {
    long long n = (long long)H * (KER[j] * H);
    transpose_w_kernel<<<(unsigned)((n + 255) / 256), 256, 0, stream>>>(
        convW[j], convWt[j], KER[j] * H, H, H);
  }
  {
    long long n = (long long)HH * H;
    transpose_w_kernel<<<(unsigned)((n + 255) / 256), 256, 0, stream>>>(
        attn_w1, attnWt, H, HH, HH);
  }
  {
    long long n = (long long)NPAD_CLS * H;
    transpose_w_kernel<<<(unsigned)((n + 255) / 256), 256, 0, stream>>>(
        cls_w, clsWt, H, TYPESP1, NPAD_CLS);
  }

  // 3) Conv stack as overlapped-row WMMA GEMMs (sequential on stream).
  for (int j = 0; j < NKER; ++j) {
    GemmArgs g{};
    g.A      = (j == 0) ? xbf : (feats + (long long)(j - 1) * PLANE);
    g.Bw     = convWt[j];
    g.bias   = convB[j];
    g.outB   = feats + (long long)j * PLANE;
    g.outF   = (j == 0) ? out_queries : nullptr;   // queries = gelu(conv1 out)
    g.M      = NB * (LSEQ - OFFS[j]);
    g.Npad   = H; g.Nout = H;
    g.K      = KER[j] * H;
    g.lda    = H;
    g.Lvalid = LSEQ - OFFS[j]; g.Ltot = LSEQ;
    g.ldOutB = H; g.ldOutF = H;
    g.act    = 1;
    long long tiles = (long long)((g.M + 16 * WM - 1) / (16 * WM)) *
                      ((g.Npad + 16 * WN - 1) / (16 * WN));
    wmma_gemm_kernel<<<(unsigned)((tiles + 7) / 8), 256, 0, stream>>>(g);
  }

  // 4) h1 = gelu(features @ attn_w1 + b1)  (M = 4*B*L, N = 256, K = 512)
  {
    GemmArgs g{};
    g.A = feats; g.Bw = attnWt; g.bias = attn_b1;
    g.outB = h1; g.outF = nullptr;
    g.M = NKER * NB * LSEQ; g.Npad = HH; g.Nout = HH; g.K = H;
    g.lda = H; g.Lvalid = 0; g.Ltot = 0;
    g.ldOutB = HH; g.ldOutF = 0; g.act = 1;
    long long tiles = (long long)((g.M + 16 * WM - 1) / (16 * WM)) *
                      ((g.Npad + 16 * WN - 1) / (16 * WN));
    wmma_gemm_kernel<<<(unsigned)((tiles + 7) / 8), 256, 0, stream>>>(g);
  }

  // 5) queries_logits = feats[0] @ cls_w + cls_b  (N = 101, padded to 112)
  {
    GemmArgs g{};
    g.A = feats; g.Bw = clsWt; g.bias = cls_b;
    g.outB = nullptr; g.outF = out_logits;
    g.M = NB * LSEQ; g.Npad = NPAD_CLS; g.Nout = TYPESP1; g.K = H;
    g.lda = H; g.Lvalid = 0; g.Ltot = 0;
    g.ldOutB = 0; g.ldOutF = TYPESP1; g.act = 0;
    long long tiles = (long long)((g.M + 16 * WM - 1) / (16 * WM)) *
                      ((g.Npad + 16 * WN - 1) / (16 * WN));
    wmma_gemm_kernel<<<(unsigned)((tiles + 7) / 8), 256, 0, stream>>>(g);
  }

  // 6) scores = h1 @ attn_w2 + b2   (N=2; w2 staged via TDM)
  {
    int Mrows = NKER * NB * LSEQ;
    scores_kernel<<<(Mrows + 255) / 256, 256, 0, stream>>>(h1, attn_w2, attn_b2,
                                                           scbuf, Mrows);
  }

  // 7) gather + softmax + weighted span endpoints -> queries_locs
  {
    int n = NB * LSEQ;
    gather_kernel<<<(n + 255) / 256, 256, 0, stream>>>(scbuf, masks, sel,
                                                       out_locs, smax);
  }
}
